// GraphTransformerLayer_12987981103702
// MI455X (gfx1250) — matrix operand order repaired and checked by reading, not compiled, not run on hardware
//
#include <hip/hip_runtime.h>
#include <hip/hip_bf16.h>

typedef __attribute__((ext_vector_type(16))) _Float16 v16h;
typedef __attribute__((ext_vector_type(8)))  _Float16 v8h;
typedef __attribute__((ext_vector_type(8)))  float    v8f;
typedef __attribute__((ext_vector_type(4)))  float    v4f;

#define NN    50000
#define NE    800000
#define DIMM  128
#define NH    8

// workspace layout (float offsets)
#define WS_Q     0
#define WS_K     6400000
#define WS_V     12800000
#define WS_WV    19200000
#define WS_Z     25600000   // 400000 floats
#define WS_STAT  26000000   // 1024 floats: bn1h(sum,sq) bn1e(sum,sq) bn2h(sum,sq) bn2e(sum,sq)
#define WS_SCSH  26001024   // 1024 floats: [scale,shift] x {1h,1e,2h,2e}
#define WS_PACK  26002048   // packed f16 weights (229376 halves)

// packed half offsets
#define PK_WQ   0
#define PK_WK   16384
#define PK_WV   32768
#define PK_WE   49152
#define PK_WOH  65536
#define PK_WOE  81920
#define PK_WH1  98304
#define PK_WH2  131072
#define PK_WE1  163840
#define PK_WE2  196608

// ---------------- WMMA helpers ----------------

__device__ __forceinline__ v8f wmma_f16(v16h a, v16h b, v8f c) {
  // v_wmma_f32_16x16x32_f16: D = A(16x32) * B(32x16) + C(16x16 f32)
  return __builtin_amdgcn_wmma_f32_16x16x32_f16(false, a, false, b, (short)0, c, false, false);
}

// A-fragment (16x32 f16) from LDS row-major tile [16][stride].
// Lane l: m = l&15, khalf = l>>4. Element j: k = kb*32 + (j<8 ? j : j+8) + khalf*8
// -> two contiguous 8-half (16B) chunks at kbase and kbase+16.
__device__ __forceinline__ v16h lds_a_frag(const _Float16* sA, int stride, int kb, int lane) {
  const int m = lane & 15, kh = lane >> 4;
  const _Float16* p = sA + m * stride + kb * 32 + kh * 8;
  v8h lo = *(const v8h*)(p);
  v8h hi = *(const v8h*)(p + 16);
  v16h a;
#pragma unroll
  for (int i = 0; i < 8; ++i) { a[i] = lo[i]; a[i + 8] = hi[i]; }
  return a;
}

// B-fragment from pre-packed weights: 16 contiguous halves (32B) per lane.
__device__ __forceinline__ v16h load_b_frag(const _Float16* pW, int nt, int KB, int kb, int lane) {
  return *(const v16h*)(pW + (size_t)((((nt * KB) + kb) * 32 + lane) << 4));
}

// ---------------- utility kernels ----------------

// Pack W[K][Nout] (f32 row-major) into WMMA B-fragment order (f16).
// B layout: lane l holds column n = nt*16 + (l&15); element j: k = kb*32 + (j<8 ? j : j+8) + (l>>4)*8.
__global__ void __launch_bounds__(256)
pack_weight_kernel(const float* __restrict__ W, _Float16* __restrict__ out, int K, int Nout) {
  const int idx = blockIdx.x * 256 + threadIdx.x;
  if (idx >= K * Nout) return;
  const int j    = idx & 15;
  const int lane = (idx >> 4) & 31;
  const int rest = idx >> 9;
  const int KB   = K >> 5;
  const int kb   = rest % KB;
  const int nt   = rest / KB;
  const int k = kb * 32 + (j < 8 ? j : j + 8) + (lane >> 4) * 8;
  const int n = nt * 16 + (lane & 15);
  out[idx] = (_Float16)W[k * Nout + n];
}

__global__ void __launch_bounds__(256)
fill_zero_kernel(float* __restrict__ p, int n) {
  const int i = blockIdx.x * 256 + threadIdx.x;
  if (i < n) p[i] = 0.f;
}

// ---------------- K1: Q,K,V projections ----------------

__global__ void __launch_bounds__(256)
qkv_kernel(const float* __restrict__ h, const _Float16* __restrict__ pk,
           const float* __restrict__ bQ, const float* __restrict__ bK, const float* __restrict__ bV,
           float* __restrict__ Qf, float* __restrict__ Kf, float* __restrict__ Vf) {
  __shared__ __attribute__((aligned(16))) _Float16 sA[16 * 136];
  const int b0 = blockIdx.x << 4;
  const int t = threadIdx.x, lane = t & 31, wave = t >> 5, kh = lane >> 4;
  {
    const int r = t >> 4, c = (t & 15) << 3;
    const float* sp = h + (size_t)(b0 + r) * DIMM + c;
    v4f x0 = *(const v4f*)sp, x1 = *(const v4f*)(sp + 4);
    _Float16* d = sA + r * 136 + c;
#pragma unroll
    for (int i = 0; i < 4; ++i) { d[i] = (_Float16)x0[i]; d[4 + i] = (_Float16)x1[i]; }
  }
  __syncthreads();
  v16h a0 = lds_a_frag(sA, 136, 0, lane), a1 = lds_a_frag(sA, 136, 1, lane),
       a2 = lds_a_frag(sA, 136, 2, lane), a3 = lds_a_frag(sA, 136, 3, lane);
  const int ncol = (wave << 4) + (lane & 15);
#pragma unroll 1
  for (int which = 0; which < 3; ++which) {
    const _Float16* pw = pk + which * 16384;
    const float* bias  = (which == 0) ? bQ : (which == 1 ? bK : bV);
    float* outp        = (which == 0) ? Qf : (which == 1 ? Kf : Vf);
    const float bv = bias[ncol];
    v8f acc;
#pragma unroll
    for (int i = 0; i < 8; ++i) acc[i] = bv;
    acc = wmma_f16(a0, load_b_frag(pw, wave, 4, 0, lane), acc);
    acc = wmma_f16(a1, load_b_frag(pw, wave, 4, 1, lane), acc);
    acc = wmma_f16(a2, load_b_frag(pw, wave, 4, 2, lane), acc);
    acc = wmma_f16(a3, load_b_frag(pw, wave, 4, 3, lane), acc);
#pragma unroll
    for (int v = 0; v < 8; ++v)
      outp[(size_t)(b0 + v + (kh << 3)) * DIMM + ncol] = acc[v];
  }
}

// ---------------- K2: fused edge kernel ----------------
// Per 16 edges: Ee=e@WE+bE (WMMA); score=K[src]*Q[dst]*Ee/4; logits->s=exp(clip);
// scatter wV,z atomics; e1_pre = e + score@WOe + bOe -> d_out e-region; BN1e stats.

__global__ void __launch_bounds__(256)
edge_attn_kernel(const float* __restrict__ e, const int* __restrict__ src, const int* __restrict__ dst,
                 const float* __restrict__ Qf, const float* __restrict__ Kf, const float* __restrict__ Vf,
                 const _Float16* __restrict__ pWE, const float* __restrict__ bE,
                 const _Float16* __restrict__ pWOe, const float* __restrict__ bOe,
                 float* __restrict__ wV, float* __restrict__ z, float* __restrict__ e1_out,
                 float* __restrict__ bnSum, float* __restrict__ bnSq) {
  __shared__ __attribute__((aligned(16))) _Float16 sE[16 * 136];
  __shared__ __attribute__((aligned(16))) _Float16 sAttn[16 * 136];
  __shared__ float sEf[16 * DIMM];
  __shared__ float sS[16 * NH];
  __shared__ int sSrc[16];
  __shared__ int sDst[16];
  const int b0 = blockIdx.x << 4;
  const int t = threadIdx.x, lane = t & 31, wave = t >> 5, kh = lane >> 4;
  if (t < 16) { sSrc[t] = src[b0 + t]; sDst[t] = dst[b0 + t]; }
  {
    const int r = t >> 4, c = (t & 15) << 3;
    const float* sp = e + (size_t)(b0 + r) * DIMM + c;
    v4f x0 = *(const v4f*)sp, x1 = *(const v4f*)(sp + 4);
    float* df = sEf + r * DIMM + c;
    _Float16* dh = sE + r * 136 + c;
#pragma unroll
    for (int i = 0; i < 4; ++i) {
      df[i] = x0[i];  df[4 + i] = x1[i];
      dh[i] = (_Float16)x0[i]; dh[4 + i] = (_Float16)x1[i];
    }
  }
  __syncthreads();
  const int ncol = (wave << 4) + (lane & 15);   // this wave's tile == head `wave`
  // Ee tile
  v8f cE;
  { const float bv = bE[ncol];
#pragma unroll
    for (int i = 0; i < 8; ++i) cE[i] = bv; }
#pragma unroll
  for (int kb = 0; kb < 4; ++kb)
    cE = wmma_f16(lds_a_frag(sE, 136, kb, lane), load_b_frag(pWE, wave, 4, kb, lane), cE);
  // score + logits + e_attn
  float sc[8];
#pragma unroll
  for (int v = 0; v < 8; ++v) {
    const int m = v + (kh << 3);
    const int srow = sSrc[m], drow = sDst[m];
    sc[v] = Kf[(size_t)srow * DIMM + ncol] * Qf[(size_t)drow * DIMM + ncol] * 0.25f * cE[v];
  }
#pragma unroll
  for (int v = 0; v < 8; ++v) {
    const int m = v + (kh << 3);
    float r = sc[v];
    r += __shfl_xor(r, 1); r += __shfl_xor(r, 2);
    r += __shfl_xor(r, 4); r += __shfl_xor(r, 8);
    const float sexp = __expf(fminf(fmaxf(r, -5.f), 5.f));
    if ((lane & 15) == 0) sS[m * NH + wave] = sexp;
    sAttn[m * 136 + ncol] = (_Float16)sc[v];
  }
  __syncthreads();
  // scatter V[src]*s -> wV[dst], s -> z[dst]
  {
    const int m = t >> 4, c = (t & 15) << 3, head = c >> 4;
    const int srow = sSrc[m], drow = sDst[m];
    const float sv = sS[m * NH + head];
    const float* vp = Vf + (size_t)srow * DIMM + c;
    v4f x0 = *(const v4f*)vp, x1 = *(const v4f*)(vp + 4);
    float* wp = wV + (size_t)drow * DIMM + c;
#pragma unroll
    for (int i = 0; i < 4; ++i) {
      atomicAdd(wp + i,     x0[i] * sv);
      atomicAdd(wp + 4 + i, x1[i] * sv);
    }
  }
  if (t < 16 * NH) {
    const int m = t >> 3, head = t & 7;
    atomicAdd(&z[(size_t)sDst[m] * NH + head], sS[m * NH + head]);
  }
  // e1 = e + e_attn @ WOe + bOe
  v8f cO;
  { const float bv = bOe[ncol];
#pragma unroll
    for (int i = 0; i < 8; ++i) cO[i] = bv; }
#pragma unroll
  for (int kb = 0; kb < 4; ++kb)
    cO = wmma_f16(lds_a_frag(sAttn, 136, kb, lane), load_b_frag(pWOe, wave, 4, kb, lane), cO);
  float csum = 0.f, csq = 0.f;
#pragma unroll
  for (int v = 0; v < 8; ++v) {
    const int m = v + (kh << 3);
    const float val = sEf[m * DIMM + ncol] + cO[v];
    e1_out[(size_t)(b0 + m) * DIMM + ncol] = val;
    csum += val; csq += val * val;
  }
  csum += __shfl_xor(csum, 16);
  csq  += __shfl_xor(csq, 16);
  if (lane < 16) { atomicAdd(&bnSum[ncol], csum); atomicAdd(&bnSq[ncol], csq); }
}

// ---------------- K3: node attention output + WOh ----------------

__global__ void __launch_bounds__(256)
node_attn_kernel(const float* __restrict__ h,
                 const float* __restrict__ wV, const float* __restrict__ z,
                 const _Float16* __restrict__ pWOh, const float* __restrict__ bOh,
                 float* __restrict__ h1_out,
                 float* __restrict__ bnSum, float* __restrict__ bnSq) {
  __shared__ __attribute__((aligned(16))) _Float16 sA[16 * 136];
  const int b0 = blockIdx.x << 4;
  const int t = threadIdx.x, lane = t & 31, wave = t >> 5, kh = lane >> 4;
  {
    const int r = t >> 4, c = (t & 15) << 3, head = c >> 4;
    const int node = b0 + r;
    const float rz = 1.f / (z[(size_t)node * NH + head] + 1e-6f);
    const float* wp = wV + (size_t)node * DIMM + c;
    v4f x0 = *(const v4f*)wp, x1 = *(const v4f*)(wp + 4);
    _Float16* d = sA + r * 136 + c;
#pragma unroll
    for (int i = 0; i < 4; ++i) { d[i] = (_Float16)(x0[i] * rz); d[4 + i] = (_Float16)(x1[i] * rz); }
  }
  __syncthreads();
  const int ncol = (wave << 4) + (lane & 15);
  v8f acc;
  { const float bv = bOh[ncol];
#pragma unroll
    for (int i = 0; i < 8; ++i) acc[i] = bv; }
#pragma unroll
  for (int kb = 0; kb < 4; ++kb)
    acc = wmma_f16(lds_a_frag(sA, 136, kb, lane), load_b_frag(pWOh, wave, 4, kb, lane), acc);
  float csum = 0.f, csq = 0.f;
#pragma unroll
  for (int v = 0; v < 8; ++v) {
    const int m = v + (kh << 3);
    const float val = h[(size_t)(b0 + m) * DIMM + ncol] + acc[v];
    h1_out[(size_t)(b0 + m) * DIMM + ncol] = val;
    csum += val; csq += val * val;
  }
  csum += __shfl_xor(csum, 16);
  csq  += __shfl_xor(csq, 16);
  if (lane < 16) { atomicAdd(&bnSum[ncol], csum); atomicAdd(&bnSq[ncol], csq); }
}

// ---------------- BN finalize: (sum,sumsq) -> (scale,shift) ----------------

__global__ void
bn_finalize_kernel(const float* __restrict__ sums, float* __restrict__ scsh,
                   const float* __restrict__ gh, const float* __restrict__ bh,
                   const float* __restrict__ ge, const float* __restrict__ be,
                   float cntH, float cntE) {
  const int t = threadIdx.x;
  if (t >= 256) return;
  const int c = t & 127;
  const bool is_e = t >= 128;
  const float cnt = is_e ? cntE : cntH;
  const float* S = sums + (is_e ? 256 : 0);
  const float mu  = S[c] / cnt;
  const float var = S[128 + c] / cnt - mu * mu;
  const float g = (is_e ? ge : gh)[c];
  const float b = (is_e ? be : bh)[c];
  const float sc = g * rsqrtf(var + 1e-5f);
  float* O = scsh + (is_e ? 256 : 0);
  O[c] = sc;
  O[128 + c] = b - mu * sc;
}

// ---------------- fused FFN: y = xbn + relu(xbn@W1+b1)@W2+b2 (in place) ----------------

__global__ void __launch_bounds__(256)
ffn_kernel(float* __restrict__ xio,
           const float* __restrict__ scale, const float* __restrict__ shift,
           const _Float16* __restrict__ pW1, const float* __restrict__ b1,
           const _Float16* __restrict__ pW2, const float* __restrict__ b2,
           float* __restrict__ bnSum, float* __restrict__ bnSq) {
  __shared__ __attribute__((aligned(16))) _Float16 sX[16 * 136];
  __shared__ __attribute__((aligned(16))) _Float16 sT[16 * 264];
  __shared__ float sXf[16 * DIMM];
  const int b0 = blockIdx.x << 4;
  const int t = threadIdx.x, lane = t & 31, wave = t >> 5, kh = lane >> 4;
  {
    const int r = t >> 4, c = (t & 15) << 3;
    const float* sp = xio + (size_t)(b0 + r) * DIMM + c;
    v4f x0 = *(const v4f*)sp, x1 = *(const v4f*)(sp + 4);
    float* df = sXf + r * DIMM + c;
    _Float16* dh = sX + r * 136 + c;
#pragma unroll
    for (int i = 0; i < 4; ++i) {
      const float a = x0[i] * scale[c + i] + shift[c + i];
      const float b = x1[i] * scale[c + 4 + i] + shift[c + 4 + i];
      df[i] = a; df[4 + i] = b;
      dh[i] = (_Float16)a; dh[4 + i] = (_Float16)b;
    }
  }
  __syncthreads();
  // GEMM1: 16x128 @ 128x256 -> relu -> sT (f16)
  v16h a0 = lds_a_frag(sX, 136, 0, lane), a1 = lds_a_frag(sX, 136, 1, lane),
       a2 = lds_a_frag(sX, 136, 2, lane), a3 = lds_a_frag(sX, 136, 3, lane);
#pragma unroll
  for (int hlf = 0; hlf < 2; ++hlf) {
    const int nt = wave + (hlf << 3);
    const int nc = (nt << 4) + (lane & 15);
    const float bv = b1[nc];
    v8f acc;
#pragma unroll
    for (int i = 0; i < 8; ++i) acc[i] = bv;
    acc = wmma_f16(a0, load_b_frag(pW1, nt, 4, 0, lane), acc);
    acc = wmma_f16(a1, load_b_frag(pW1, nt, 4, 1, lane), acc);
    acc = wmma_f16(a2, load_b_frag(pW1, nt, 4, 2, lane), acc);
    acc = wmma_f16(a3, load_b_frag(pW1, nt, 4, 3, lane), acc);
#pragma unroll
    for (int v = 0; v < 8; ++v)
      sT[(v + (kh << 3)) * 264 + nc] = (_Float16)fmaxf(acc[v], 0.f);
  }
  __syncthreads();
  // GEMM2: 16x256 @ 256x128 + residual
  const int ncol = (wave << 4) + (lane & 15);
  v8f c2;
  { const float bv = b2[ncol];
#pragma unroll
    for (int i = 0; i < 8; ++i) c2[i] = bv; }
#pragma unroll
  for (int kb = 0; kb < 8; ++kb)
    c2 = wmma_f16(lds_a_frag(sT, 264, kb, lane), load_b_frag(pW2, wave, 8, kb, lane), c2);
  float csum = 0.f, csq = 0.f;
#pragma unroll
  for (int v = 0; v < 8; ++v) {
    const int m = v + (kh << 3);
    const float val = sXf[m * DIMM + ncol] + c2[v];
    xio[(size_t)(b0 + m) * DIMM + ncol] = val;
    csum += val; csq += val * val;
  }
  csum += __shfl_xor(csum, 16);
  csq  += __shfl_xor(csq, 16);
  if (lane < 16) { atomicAdd(&bnSum[ncol], csum); atomicAdd(&bnSq[ncol], csq); }
}

// ---------------- BN apply (in place) ----------------

__global__ void __launch_bounds__(256)
bn_apply_kernel(float* __restrict__ x, const float* __restrict__ scale,
                const float* __restrict__ shift, size_t total) {
  const size_t i = (size_t)blockIdx.x * 256 + threadIdx.x;
  if (i < total) {
    const int c = (int)(i & 127);
    x[i] = x[i] * scale[c] + shift[c];
  }
}

// ---------------- host launcher ----------------

extern "C" void kernel_launch(void* const* d_in, const int* in_sizes, int n_in,
                              void* d_out, int out_size, void* d_ws, size_t ws_size,
                              hipStream_t stream) {
  (void)in_sizes; (void)n_in; (void)out_size; (void)ws_size;
  const float* h   = (const float*)d_in[0];
  const float* e   = (const float*)d_in[1];
  const int*   src = (const int*)d_in[2];
  const int*   dst = (const int*)d_in[3];
  const float* WQ  = (const float*)d_in[4];  const float* bQ  = (const float*)d_in[5];
  const float* WK  = (const float*)d_in[6];  const float* bK  = (const float*)d_in[7];
  const float* WV  = (const float*)d_in[8];  const float* bV  = (const float*)d_in[9];
  const float* WE  = (const float*)d_in[10]; const float* bE  = (const float*)d_in[11];
  const float* WOh = (const float*)d_in[12]; const float* bOh = (const float*)d_in[13];
  const float* WOe = (const float*)d_in[14]; const float* bOe = (const float*)d_in[15];
  const float* Wh1 = (const float*)d_in[16]; const float* bh1 = (const float*)d_in[17];
  const float* Wh2 = (const float*)d_in[18]; const float* bh2 = (const float*)d_in[19];
  const float* We1 = (const float*)d_in[20]; const float* be1 = (const float*)d_in[21];
  const float* We2 = (const float*)d_in[22]; const float* be2 = (const float*)d_in[23];
  const float* g1h = (const float*)d_in[24]; const float* g1e = (const float*)d_in[25];
  const float* g2h = (const float*)d_in[26]; const float* g2e = (const float*)d_in[27];
  const float* b1h = (const float*)d_in[28]; const float* b1e = (const float*)d_in[29];
  const float* b2h = (const float*)d_in[30]; const float* b2e = (const float*)d_in[31];

  float* ws = (float*)d_ws;
  float* Qf = ws + WS_Q;
  float* Kf = ws + WS_K;
  float* Vf = ws + WS_V;
  float* wV = ws + WS_WV;
  float* zb = ws + WS_Z;
  float* st = ws + WS_STAT;
  float* ss = ws + WS_SCSH;
  _Float16* pk = (_Float16*)(ws + WS_PACK);

  float* outH = (float*)d_out;                    // h2: 50000 x 128
  float* outE = outH + (size_t)NN * DIMM;         // e2: 800000 x 128

  // pack weights into WMMA B-fragment order (tiny; L2-resident afterwards)
  pack_weight_kernel<<<64, 256, 0, stream>>>(WQ,  pk + PK_WQ,  128, 128);
  pack_weight_kernel<<<64, 256, 0, stream>>>(WK,  pk + PK_WK,  128, 128);
  pack_weight_kernel<<<64, 256, 0, stream>>>(WV,  pk + PK_WV,  128, 128);
  pack_weight_kernel<<<64, 256, 0, stream>>>(WE,  pk + PK_WE,  128, 128);
  pack_weight_kernel<<<64, 256, 0, stream>>>(WOh, pk + PK_WOH, 128, 128);
  pack_weight_kernel<<<64, 256, 0, stream>>>(WOe, pk + PK_WOE, 128, 128);
  pack_weight_kernel<<<128, 256, 0, stream>>>(Wh1, pk + PK_WH1, 128, 256);
  pack_weight_kernel<<<128, 256, 0, stream>>>(Wh2, pk + PK_WH2, 256, 128);
  pack_weight_kernel<<<128, 256, 0, stream>>>(We1, pk + PK_WE1, 128, 256);
  pack_weight_kernel<<<128, 256, 0, stream>>>(We2, pk + PK_WE2, 256, 128);

  // zero wV, z and BN stat accumulators (contiguous region)
  fill_zero_kernel<<<(6801024 + 255) / 256, 256, 0, stream>>>(ws + WS_WV, 6801024);

  // Q,K,V
  qkv_kernel<<<NN / 16, 256, 0, stream>>>(h, pk, bQ, bK, bV, Qf, Kf, Vf);

  // fused edge attention (e1_pre -> d_out e-region, BN1e stats, wV/z atomics)
  edge_attn_kernel<<<NE / 16, 256, 0, stream>>>(e, src, dst, Qf, Kf, Vf,
                                                pk + PK_WE, bE, pk + PK_WOE, bOe,
                                                wV, zb, outE, st + 256, st + 384);

  // node attention output + WOh (h1_pre -> d_out h-region, BN1h stats)
  node_attn_kernel<<<NN / 16, 256, 0, stream>>>(h, wV, zb, pk + PK_WOH, bOh,
                                                outH, st + 0, st + 128);

  // BN1 scale/shift
  bn_finalize_kernel<<<1, 256, 0, stream>>>(st, ss, g1h, b1h, g1e, b1e,
                                            (float)NN, (float)NE);

  // fused FFNs (BN1 applied inline, result overwrites d_out, BN2 stats)
  ffn_kernel<<<NN / 16, 256, 0, stream>>>(outH, ss + 0, ss + 128,
                                          pk + PK_WH1, bh1, pk + PK_WH2, bh2,
                                          st + 512, st + 640);
  ffn_kernel<<<NE / 16, 256, 0, stream>>>(outE, ss + 256, ss + 384,
                                          pk + PK_WE1, be1, pk + PK_WE2, be2,
                                          st + 768, st + 896);

  // BN2 scale/shift, then apply in place
  bn_finalize_kernel<<<1, 256, 0, stream>>>(st + 512, ss + 512, g2h, b2h, g2e, b2e,
                                            (float)NN, (float)NE);
  bn_apply_kernel<<<(6400000 + 255) / 256, 256, 0, stream>>>(outH, ss + 512, ss + 640,
                                                             (size_t)6400000);
  bn_apply_kernel<<<(102400000 + 255) / 256, 256, 0, stream>>>(outE, ss + 768, ss + 896,
                                                               (size_t)102400000);
}
